// GATv2_17257178595619
// MI455X (gfx1250) — matrix operand order repaired and checked by reading, not compiled, not run on hardware
//
#include <hip/hip_runtime.h>
#include <hip/hip_bf16.h>
#include <cmath>

typedef __attribute__((ext_vector_type(8)))  __bf16 v8bf;
typedef __attribute__((ext_vector_type(16))) __bf16 v16bf;
typedef __attribute__((ext_vector_type(8)))  float  v8f;

#define HIDDEN 128
#define HEADS  4
#define NEG_SLOPE 0.2f
#define NUM_GRAPHS 64

// ---------- helpers ----------
__device__ __forceinline__ unsigned short f2bf(float f) {
    unsigned u = __float_as_uint(f);
    unsigned r = u + 0x7FFFu + ((u >> 16) & 1u);   // round-to-nearest-even
    return (unsigned short)(r >> 16);
}

// float atomic max via ordered-int trick (works for mixed signs, init = -inf)
__device__ __forceinline__ void atomicMaxF(float* addr, float v) {
    if (v >= 0.0f) atomicMax((int*)addr, __float_as_int(v));
    else           atomicMin((unsigned int*)addr, __float_as_uint(v));
}

// ---------- elementwise ----------
__global__ void k_f32_to_bf16(const float* __restrict__ in,
                              unsigned short* __restrict__ out, long long n) {
    long long i = (long long)blockIdx.x * blockDim.x + threadIdx.x;
    if (i < n) out[i] = f2bf(in[i]);
}

__global__ void k_fill(float* __restrict__ p, float v, long long n) {
    long long i = (long long)blockIdx.x * blockDim.x + threadIdx.x;
    if (i < n) p[i] = v;
}

// ---------- WMMA GEMM: out[M][128] = A[M][K](bf16) @ W[128][K](bf16)^T + bias ----------
// block = 256 threads = 8 waves; wave w computes the 16x16 tile at (blockIdx.x*16, w*16)
__global__ void k_gemm_bf16(const unsigned short* __restrict__ A,
                            const unsigned short* __restrict__ W,
                            const float* __restrict__ bias,
                            float* __restrict__ out, int M, int K) {
    const int lane = threadIdx.x & 31;
    const int wv   = threadIdx.x >> 5;         // n-tile 0..7
    const int m0   = blockIdx.x * 16;
    const int n0   = wv * 16;

    int mr = m0 + (lane & 15); if (mr >= M) mr = M - 1;  // clamp: keep EXEC all-ones
    const int kbA = (lane & 16) ? 16 : 0;                // A frag K base (lanes 16-31 -> K 16..31)
    const int nc  = n0 + (lane & 15);
    const int kbB = (lane & 16) ? 16 : 0;                // B frag K base (lanes 16-31 -> K 16..31)

    const unsigned short* ap = A + (size_t)mr * K + kbA;
    const unsigned short* wp = W + (size_t)nc * K + kbB;

    v8f c = {};
    for (int k = 0; k < K; k += 32) {
        __builtin_prefetch(ap + k + 64, 0, 0);           // global_prefetch_b8
        v8bf  alo = *(const v8bf*)(ap + k);              // K = kbA+k .. +7
        v8bf  ahi = *(const v8bf*)(ap + k + 8);          // K = kbA+k+8 .. +15
        v16bf a   = __builtin_shufflevector(alo, ahi,
                        0,1,2,3,4,5,6,7,8,9,10,11,12,13,14,15);
        v16bf b   = *(const v16bf*)(wp + k);             // 16 contiguous K of W row nc
        c = __builtin_amdgcn_wmma_f32_16x16x32_bf16(
                false, a, false, b, (short)0, c, false, false);
    }

    const float bv = bias[nc];
    const int   mb = m0 + ((lane & 16) ? 8 : 0);
#pragma unroll
    for (int r = 0; r < 8; ++r) {
        int m = mb + r;
        if (m < M) out[(size_t)m * HIDDEN + nc] = c[r] + bv;
    }
}

// ---------- edge phase (wave32 per edge; lane = 4 channels; head = lane/8) ----------
__global__ void k_edge_alpha(const float* __restrict__ xl, const float* __restrict__ xr,
                             const long long* __restrict__ ei, long long E, long long ET,
                             const float* __restrict__ att,
                             float* __restrict__ alpha, float* __restrict__ amax) {
    long long e = (long long)blockIdx.x * 8 + (threadIdx.x >> 5);
    if (e >= ET) return;
    int lane = threadIdx.x & 31;
    long long s, d;
    if (e < E) { s = ei[e]; d = ei[E + e]; } else { s = e - E; d = s; }

    float4 xa = ((const float4*)(xl + s * HIDDEN))[lane];
    float4 xb = ((const float4*)(xr + d * HIDDEN))[lane];
    float4 at = ((const float4*)att)[lane];

    float p = 0.0f, t;
    t = xa.x + xb.x; t = (t > 0.0f) ? t : NEG_SLOPE * t; p += t * at.x;
    t = xa.y + xb.y; t = (t > 0.0f) ? t : NEG_SLOPE * t; p += t * at.y;
    t = xa.z + xb.z; t = (t > 0.0f) ? t : NEG_SLOPE * t; p += t * at.z;
    t = xa.w + xb.w; t = (t > 0.0f) ? t : NEG_SLOPE * t; p += t * at.w;

    p += __shfl_xor(p, 1, 32);
    p += __shfl_xor(p, 2, 32);
    p += __shfl_xor(p, 4, 32);

    if ((lane & 7) == 0) {
        int h = lane >> 3;
        alpha[e * HEADS + h] = p;
        atomicMaxF(amax + d * HEADS + h, p);
    }
}

__global__ void k_edge_exp(const long long* __restrict__ ei, long long E, long long ET,
                           const float* __restrict__ amax,
                           float* __restrict__ alpha, float* __restrict__ denom) {
    long long i = (long long)blockIdx.x * blockDim.x + threadIdx.x;
    if (i >= ET * HEADS) return;
    long long e = i >> 2; int h = (int)(i & 3);
    long long d = (e < E) ? ei[E + e] : (e - E);
    float ex = __expf(alpha[i] - amax[d * HEADS + h]);
    alpha[i] = ex;
    atomicAdd(denom + d * HEADS + h, ex);
}

__global__ void k_edge_agg(const float* __restrict__ xl,
                           const long long* __restrict__ ei, long long E, long long ET,
                           const float* __restrict__ alpha, const float* __restrict__ denom,
                           float* __restrict__ agg) {
    long long e = (long long)blockIdx.x * 8 + (threadIdx.x >> 5);
    if (e >= ET) return;
    int lane = threadIdx.x & 31;
    long long s, d;
    if (e < E) { s = ei[e]; d = ei[E + e]; } else { s = e - E; d = s; }
    int h = lane >> 3;
    float a = alpha[e * HEADS + h] / (denom[d * HEADS + h] + 1e-16f);
    float4 xv = ((const float4*)(xl + s * HIDDEN))[lane];
    float* o = agg + d * HIDDEN + lane * 4;
    atomicAdd(o + 0, xv.x * a);
    atomicAdd(o + 1, xv.y * a);
    atomicAdd(o + 2, xv.z * a);
    atomicAdd(o + 3, xv.w * a);
}

// layer-1 epilogue: h = relu(agg + bias); emit bf16 for layer-2 GEMMs
__global__ void k_post_relu_bf16(const float* __restrict__ agg, const float* __restrict__ bias,
                                 unsigned short* __restrict__ hbf, long long n) {
    long long i = (long long)blockIdx.x * blockDim.x + threadIdx.x;
    if (i < n) {
        int f = (int)(i & (HIDDEN - 1));
        float v = agg[i] + bias[f];
        v = (v > 0.0f) ? v : 0.0f;
        hbf[i] = f2bf(v);
    }
}

// layer-2 epilogue fused with global_mean_pool numerator
__global__ void k_post_pool(const float* __restrict__ agg, const float* __restrict__ bias,
                            const long long* __restrict__ batch,
                            float* __restrict__ pooled, long long n) {
    long long i = (long long)blockIdx.x * blockDim.x + threadIdx.x;
    if (i < n) {
        int f = (int)(i & (HIDDEN - 1));
        long long node = i >> 7;
        float v = agg[i] + bias[f];
        atomicAdd(pooled + batch[node] * HIDDEN + f, v);
    }
}

__global__ void k_count(const long long* __restrict__ batch, float* __restrict__ cnt, long long N) {
    long long i = (long long)blockIdx.x * blockDim.x + threadIdx.x;
    if (i < N) atomicAdd(cnt + batch[i], 1.0f);
}

__global__ void k_final(const float* __restrict__ pooled, const float* __restrict__ cnt,
                        const float* __restrict__ Wlin, const float* __restrict__ blin,
                        float* __restrict__ out) {
    __shared__ float red[HIDDEN];
    int g = blockIdx.x, t = threadIdx.x;
    float c = cnt[g]; c = (c > 1.0f) ? c : 1.0f;
    red[t] = pooled[g * HIDDEN + t] / c * Wlin[t];
    __syncthreads();
    for (int s = 64; s > 0; s >>= 1) {
        if (t < s) red[t] += red[t + s];
        __syncthreads();
    }
    if (t == 0) out[g] = red[0] + blin[0];
}

// ---------- launcher ----------
extern "C" void kernel_launch(void* const* d_in, const int* in_sizes, int n_in,
                              void* d_out, int out_size, void* d_ws, size_t ws_size,
                              hipStream_t stream) {
    const float*     x     = (const float*)d_in[0];
    const long long* ei    = (const long long*)d_in[1];
    const long long* batch = (const long long*)d_in[2];
    const float *Wl1 = (const float*)d_in[3],  *bl1 = (const float*)d_in[4];
    const float *Wr1 = (const float*)d_in[5],  *br1 = (const float*)d_in[6];
    const float *att1= (const float*)d_in[7],  *bias1=(const float*)d_in[8];
    const float *Wl2 = (const float*)d_in[9],  *bl2 = (const float*)d_in[10];
    const float *Wr2 = (const float*)d_in[11], *br2 = (const float*)d_in[12];
    const float *att2= (const float*)d_in[13], *bias2=(const float*)d_in[14];
    const float *Wlin= (const float*)d_in[15], *blin =(const float*)d_in[16];
    float* out = (float*)d_out;

    const int       FIN = 256;
    const long long N   = (long long)in_sizes[0] / FIN;
    const long long E   = (long long)in_sizes[1] / 2;
    const long long ET  = E + N;

    // bump allocator on workspace
    char* ws = (char*)d_ws; size_t off = 0;
    auto alloc = [&](size_t bytes) -> void* {
        void* p = ws + off; off += (bytes + 255) & ~(size_t)255; return p;
    };
    unsigned short* xbf  = (unsigned short*)alloc((size_t)N * FIN * 2);
    unsigned short* wlb1 = (unsigned short*)alloc((size_t)HIDDEN * FIN * 2);
    unsigned short* wrb1 = (unsigned short*)alloc((size_t)HIDDEN * FIN * 2);
    unsigned short* wlb2 = (unsigned short*)alloc((size_t)HIDDEN * HIDDEN * 2);
    unsigned short* wrb2 = (unsigned short*)alloc((size_t)HIDDEN * HIDDEN * 2);
    float* xl    = (float*)alloc((size_t)N * HIDDEN * 4);
    float* xr    = (float*)alloc((size_t)N * HIDDEN * 4);
    unsigned short* hbf = (unsigned short*)alloc((size_t)N * HIDDEN * 2);
    float* alpha = (float*)alloc((size_t)ET * HEADS * 4);
    float* amax  = (float*)alloc((size_t)N * HEADS * 4);
    float* denom = (float*)alloc((size_t)N * HEADS * 4);
    float* agg   = (float*)alloc((size_t)N * HIDDEN * 4);
    float* pooled= (float*)alloc((size_t)NUM_GRAPHS * HIDDEN * 4);
    float* cnt   = (float*)alloc((size_t)NUM_GRAPHS * 4);

    auto blocks = [](long long n, int bs) { return (unsigned)((n + bs - 1) / bs); };

    // bf16 conversions
    k_f32_to_bf16<<<blocks(N * FIN, 256), 256, 0, stream>>>(x, xbf, N * FIN);
    k_f32_to_bf16<<<blocks(HIDDEN * FIN, 256), 256, 0, stream>>>(Wl1, wlb1, HIDDEN * FIN);
    k_f32_to_bf16<<<blocks(HIDDEN * FIN, 256), 256, 0, stream>>>(Wr1, wrb1, HIDDEN * FIN);
    k_f32_to_bf16<<<blocks(HIDDEN * HIDDEN, 256), 256, 0, stream>>>(Wl2, wlb2, HIDDEN * HIDDEN);
    k_f32_to_bf16<<<blocks(HIDDEN * HIDDEN, 256), 256, 0, stream>>>(Wr2, wrb2, HIDDEN * HIDDEN);

    const unsigned mt = blocks(N, 16);
    const unsigned eb = blocks(ET, 8);          // 8 edges (waves) per 256-thread block

    // ---- layer 1 ----
    k_gemm_bf16<<<mt, 256, 0, stream>>>(xbf, wlb1, bl1, xl, (int)N, FIN);
    k_gemm_bf16<<<mt, 256, 0, stream>>>(xbf, wrb1, br1, xr, (int)N, FIN);
    k_fill<<<blocks(N * HEADS, 256), 256, 0, stream>>>(amax, -INFINITY, N * HEADS);
    k_fill<<<blocks(N * HEADS, 256), 256, 0, stream>>>(denom, 0.0f, N * HEADS);
    k_fill<<<blocks(N * HIDDEN, 256), 256, 0, stream>>>(agg, 0.0f, N * HIDDEN);
    k_edge_alpha<<<eb, 256, 0, stream>>>(xl, xr, ei, E, ET, att1, alpha, amax);
    k_edge_exp<<<blocks(ET * HEADS, 256), 256, 0, stream>>>(ei, E, ET, amax, alpha, denom);
    k_edge_agg<<<eb, 256, 0, stream>>>(xl, ei, E, ET, alpha, denom, agg);
    k_post_relu_bf16<<<blocks(N * HIDDEN, 256), 256, 0, stream>>>(agg, bias1, hbf, N * HIDDEN);

    // ---- layer 2 ----
    k_gemm_bf16<<<mt, 256, 0, stream>>>(hbf, wlb2, bl2, xl, (int)N, HIDDEN);
    k_gemm_bf16<<<mt, 256, 0, stream>>>(hbf, wrb2, br2, xr, (int)N, HIDDEN);
    k_fill<<<blocks(N * HEADS, 256), 256, 0, stream>>>(amax, -INFINITY, N * HEADS);
    k_fill<<<blocks(N * HEADS, 256), 256, 0, stream>>>(denom, 0.0f, N * HEADS);
    k_fill<<<blocks(N * HIDDEN, 256), 256, 0, stream>>>(agg, 0.0f, N * HIDDEN);
    k_edge_alpha<<<eb, 256, 0, stream>>>(xl, xr, ei, E, ET, att2, alpha, amax);
    k_edge_exp<<<blocks(ET * HEADS, 256), 256, 0, stream>>>(ei, E, ET, amax, alpha, denom);
    k_edge_agg<<<eb, 256, 0, stream>>>(xl, ei, E, ET, alpha, denom, agg);

    // ---- pool + head ----
    k_fill<<<blocks(NUM_GRAPHS * HIDDEN, 256), 256, 0, stream>>>(pooled, 0.0f, NUM_GRAPHS * HIDDEN);
    k_fill<<<1, 64, 0, stream>>>(cnt, 0.0f, NUM_GRAPHS);
    k_post_pool<<<blocks(N * HIDDEN, 256), 256, 0, stream>>>(agg, bias2, batch, pooled, N * HIDDEN);
    k_count<<<blocks(N, 256), 256, 0, stream>>>(batch, cnt, N);
    k_final<<<NUM_GRAPHS, HIDDEN, 0, stream>>>(pooled, cnt, Wlin, blin, out);
}
